// MyModel_87522843559835
// MI455X (gfx1250) — matrix operand order repaired and checked by reading, not compiled, not run on hardware
//
#include <hip/hip_runtime.h>
#include <hip/hip_bf16.h>

// MI455X / gfx1250: wave32, WMMA 16x16x32 f16 -> f32 accumulate, TDM staging.
// Fused attention: W pre-swizzle -> QKV projection (TDM + WMMA) ->
//                  flash attention (WMMA + online softmax).

typedef __attribute__((ext_vector_type(16))) _Float16 v16h;
typedef __attribute__((ext_vector_type(8)))  float    v8f;
typedef __attribute__((ext_vector_type(4)))  unsigned int v4u;
typedef __attribute__((ext_vector_type(8)))  int      v8i;
typedef __attribute__((ext_vector_type(4)))  int      v4i;

#define B_ 4
#define S_ 2048
#define F_ 512
#define D_ 64

__device__ __forceinline__ v8f wmma_f16(v16h a, v16h b, v8f c) {
  // (neg_a, A, neg_b, B, c_mod, C, reuse_a, reuse_b)
  return __builtin_amdgcn_wmma_f32_16x16x32_f16(false, a, false, b, (short)0, c,
                                                false, false);
}

// ---------------------------------------------------------------------------
// Kernel 0: pack Wq/Wk/Wv (f32, FxD) into f16 fragments in the per-lane WMMA
// B-layout, so the projection kernel loads each B-fragment as one contiguous
// v16h.  Layout: Wsw[mat][kk][nt][lane][i], i=0..15, K = kk*32 + half*16 + i,
// N = nt*16 + (lane&15).
// ---------------------------------------------------------------------------
__global__ __launch_bounds__(256) void pack_w_kernel(
    const float* __restrict__ Wq, const float* __restrict__ Wk,
    const float* __restrict__ Wv, _Float16* __restrict__ Wsw)
{
  const int idx = blockIdx.x * 256 + threadIdx.x;   // 0 .. 3*16*4*32*16-1
  if (idx >= 3 * 16 * 4 * 32 * 16) return;
  const int i    = idx & 15;
  const int lane = (idx >> 4) & 31;
  const int nt   = (idx >> 9) & 3;
  const int kk   = (idx >> 11) & 15;
  const int mat  = idx >> 15;
  const float* W = (mat == 0) ? Wq : (mat == 1) ? Wk : Wv;
  const int half = lane >> 4;
  const int col  = lane & 15;
  const int k    = kk * 32 + half * 16 + i;
  Wsw[idx] = (_Float16)W[(size_t)k * D_ + nt * 16 + col];
}

// ---------------------------------------------------------------------------
// Kernel 1: QKV projection.  One block = one 16-row tile of x (B*S flat rows),
// staged into LDS by the Tensor Data Mover.  3 waves; wave w computes the
// 16x64 tile of {Q, K, V^T}[w]: 16 K-steps x 4 N-tiles of WMMA with the
// A-fragment hoisted.  V is stored transposed (B,D,S) for kernel 2.
// ---------------------------------------------------------------------------
__global__ __launch_bounds__(96) void qkv_proj_kernel(
    const float* __restrict__ x, const _Float16* __restrict__ Wsw,
    const float* __restrict__ bq, const float* __restrict__ bk,
    const float* __restrict__ bv,
    _Float16* __restrict__ Q, _Float16* __restrict__ K,
    _Float16* __restrict__ Vt)
{
  __shared__ float xf[16][F_];   // 32 KB of the 320 KB/WGP LDS

  const int tid  = threadIdx.x;
  const int row0 = blockIdx.x * 16;        // flat row base into (B*S)
  const int wave = tid >> 5;
  const int lane = tid & 31;
  const int half = lane >> 4;
  const int col  = lane & 15;
  const int m    = lane & 15;

#if __has_builtin(__builtin_amdgcn_tensor_load_to_lds)
  if (wave == 0) {
    // Tensor DMA descriptor: 2D tile, dim0 = 512 f32 elements (one row),
    // dim1 = 16 rows, tensor_dim0_stride = 512.
    const unsigned long long ga =
        (unsigned long long)(const void*)(x + (size_t)row0 * F_);
    const unsigned lds_off = (unsigned)(uintptr_t)(&xf[0][0]);
    v4u g0;
    g0[0] = 1u;                                   // count=1, user descriptor
    g0[1] = lds_off;                              // lds_addr
    g0[2] = (unsigned)(ga & 0xFFFFFFFFu);         // global_addr[31:0]
    g0[3] = (unsigned)((ga >> 32) & 0x01FFFFFFu)  // global_addr[56:32]
            | (2u << 30);                         // type = 2 (image)
    v8i g1;
    g1[0] = 0x20000;                              // data_size=2 (4B), mask=0
    g1[1] = (int)(512u << 16);                    // tensor_dim0[15:0] << 16
    g1[2] = (int)((8192u & 0xFFFFu) << 16);       // tensor_dim0 hi | dim1 lo
    g1[3] = (int)(512u << 16);                    // dim1 hi | tile_dim0=512
    g1[4] = 16;                                   // tile_dim1=16, tile_dim2=0
    g1[5] = 512;                                  // tensor_dim0_stride[31:0]
    g1[6] = 0;                                    // stride0 hi | stride1 lo
    g1[7] = 0;
    v4i z4 = {0, 0, 0, 0};
#if defined(__clang_major__) && __clang_major__ >= 23
    v8i z8 = {0, 0, 0, 0, 0, 0, 0, 0};
    __builtin_amdgcn_tensor_load_to_lds(g0, g1, z4, z4, z8, 0);
#else
    __builtin_amdgcn_tensor_load_to_lds(g0, g1, z4, z4, 0);
#endif
    __builtin_amdgcn_s_wait_tensorcnt(0);
  }
#else
  for (int idx = tid; idx < 16 * F_; idx += 96) {
    const int r = idx >> 9;
    const int f = idx & (F_ - 1);
    xf[r][f] = x[(size_t)(row0 + r) * F_ + f];
  }
#endif
  __syncthreads();

  const float* bia = (wave == 0) ? bq : (wave == 1) ? bk : bv;

  v8f acc[4];
#pragma unroll
  for (int nt = 0; nt < 4; ++nt) {
    v8f z = {0.f, 0.f, 0.f, 0.f, 0.f, 0.f, 0.f, 0.f};
    acc[nt] = z;
  }

#pragma unroll
  for (int kk = 0; kk < F_ / 32; ++kk) {
    const int k0 = kk * 32;
    // A-fragment (16x32 f16): lane<16 holds row m, K in {0..15} (+16 hi)
    const v8f xlo = *(const v8f*)&xf[m][k0 + half * 16];
    const v8f xhi = *(const v8f*)&xf[m][k0 + half * 16 + 8];
    v16h a;
#pragma unroll
    for (int i = 0; i < 8; ++i) {
      a[i]     = (_Float16)xlo[i];
      a[i + 8] = (_Float16)xhi[i];
    }
#pragma unroll
    for (int nt = 0; nt < 4; ++nt) {
      const v16h b = *(const v16h*)(
          Wsw + ((((size_t)wave * 16 + kk) * 4 + nt) * 32 + lane) * 16);
      acc[nt] = wmma_f16(a, b, acc[nt]);
    }
  }

  // C/D layout: VGPR r -> row (r + half*8), col = lane%16
  if (wave == 2) {
    const int bb = row0 / S_;
    const int s0 = row0 - bb * S_;
#pragma unroll
    for (int nt = 0; nt < 4; ++nt) {
      const int n = nt * 16 + col;
      const float bval = bia[n];
#pragma unroll
      for (int r = 0; r < 8; ++r)
        Vt[((size_t)bb * D_ + n) * S_ + s0 + r + half * 8] =
            (_Float16)(acc[nt][r] + bval);
    }
  } else {
    _Float16* Out = (wave == 0) ? Q : K;
#pragma unroll
    for (int nt = 0; nt < 4; ++nt) {
      const int n = nt * 16 + col;
      const float bval = bia[n];
#pragma unroll
      for (int r = 0; r < 8; ++r)
        Out[(size_t)(row0 + r + half * 8) * D_ + n] =
            (_Float16)(acc[nt][r] + bval);
    }
  }
}

// ---------------------------------------------------------------------------
// Kernel 2: flash attention.  One wave per 16 query rows (4 waves / block).
// Per 32-key block: 4 WMMA (QK^T over D=64), masked scaled online softmax with
// 16-lane shuffle reductions, P staged through LDS (C-layout -> A-layout
// transpose), then 4 WMMA (P @ V) using contiguous v16h loads from V^T.
// ---------------------------------------------------------------------------
__global__ __launch_bounds__(128) void flash_attn_kernel(
    const _Float16* __restrict__ Q, const _Float16* __restrict__ K,
    const _Float16* __restrict__ Vt, const int* __restrict__ mask,
    float* __restrict__ out)
{
  __shared__ _Float16 pbuf[4][16][32];     // per-wave P tile staging (4 KB)

  const int tid  = threadIdx.x;
  const int wave = tid >> 5;
  const int lane = tid & 31;
  const int half = lane >> 4;
  const int col  = lane & 15;
  const int m    = lane & 15;

  const int qtile = blockIdx.x * 4 + wave;     // 0 .. B*S/16-1
  const int qbase = qtile * 16;                // flat query row base
  const int b     = qbase / S_;

  // Resident Q A-fragments: two D-chunks of 32.
  v16h qa[2];
#pragma unroll
  for (int j = 0; j < 2; ++j) {
    const _Float16* qrow = Q + (size_t)(qbase + m) * D_ + j * 32;
#pragma unroll
    for (int i = 0; i < 8; ++i) {
      qa[j][i]     = qrow[half * 16 + i];
      qa[j][i + 8] = qrow[half * 16 + 8 + i];
    }
  }

  float m_i[8], l_i[8];
  v8f oacc[4];
#pragma unroll
  for (int r = 0; r < 8; ++r) { m_i[r] = -1e30f; l_i[r] = 0.f; }
#pragma unroll
  for (int t = 0; t < 4; ++t) {
    v8f z = {0.f, 0.f, 0.f, 0.f, 0.f, 0.f, 0.f, 0.f};
    oacc[t] = z;
  }

  const int* mrow = mask + b * S_;

  for (int kt = 0; kt < S_ / 32; ++kt) {
    const int key0 = kt * 32;

    // Prefetch next key block of K and V^T (global_prefetch_b8 path).
    if (kt + 1 < S_ / 32) {
      __builtin_prefetch(K + (size_t)(b * S_ + key0 + 32 + lane) * D_, 0, 1);
      __builtin_prefetch(Vt + ((size_t)b * D_ + lane) * S_ + key0 + 32, 0, 1);
    }

    // ---- S = Q @ K^T for two 16-key sub-tiles -----------------------------
    v8f sacc[2];
#pragma unroll
    for (int s2 = 0; s2 < 2; ++s2) {
      v8f z = {0.f, 0.f, 0.f, 0.f, 0.f, 0.f, 0.f, 0.f};
      sacc[s2] = z;
      const int keyb = key0 + s2 * 16;
#pragma unroll
      for (int j = 0; j < 2; ++j) {
        // B-fragment of K^T: column n = key (keyb+col); 16 contiguous f16.
        const v16h bk_ = *(const v16h*)(K + (size_t)(b * S_ + keyb + col) * D_ +
                                        j * 32 + half * 16);
        sacc[s2] = wmma_f16(qa[j], bk_, sacc[s2]);
      }
      // scale by 1/sqrt(D) and apply additive key mask (column = keyb+col)
      const float madd = (1.0f - (float)mrow[keyb + col]) * -1e9f;
#pragma unroll
      for (int r = 0; r < 8; ++r)
        sacc[s2][r] = sacc[s2][r] * 0.125f + madd;
    }

    // ---- online softmax (row stats reduce across 16-lane half-wave) -------
#pragma unroll
    for (int r = 0; r < 8; ++r) {
      float rmax = fmaxf(sacc[0][r], sacc[1][r]);
#pragma unroll
      for (int mm = 1; mm < 16; mm <<= 1)
        rmax = fmaxf(rmax, __shfl_xor(rmax, mm, 32));
      const float mnew  = fmaxf(m_i[r], rmax);
      const float alpha = __expf(m_i[r] - mnew);
      const float p0 = __expf(sacc[0][r] - mnew);
      const float p1 = __expf(sacc[1][r] - mnew);
      float rs = p0 + p1;
#pragma unroll
      for (int mm = 1; mm < 16; mm <<= 1)
        rs += __shfl_xor(rs, mm, 32);
      l_i[r] = l_i[r] * alpha + rs;
      m_i[r] = mnew;
#pragma unroll
      for (int t = 0; t < 4; ++t)
        oacc[t][r] *= alpha;
      // stage P in row-major LDS (C-layout coords)
      pbuf[wave][r + half * 8][col]      = (_Float16)p0;
      pbuf[wave][r + half * 8][16 + col] = (_Float16)p1;
    }

    // re-read P as an A-fragment (16x32)
    v16h pa;
#pragma unroll
    for (int i = 0; i < 8; ++i) {
      pa[i]     = pbuf[wave][m][half * 16 + i];
      pa[i + 8] = pbuf[wave][m][half * 16 + 8 + i];
    }

    // ---- O += P @ V: contiguous v16h B-fragments from V^T -----------------
#pragma unroll
    for (int t = 0; t < 4; ++t) {
      const v16h bv_ = *(const v16h*)(
          Vt + ((size_t)b * D_ + t * 16 + col) * S_ + key0 + half * 16);
      oacc[t] = wmma_f16(pa, bv_, oacc[t]);
    }
  }

  // ---- epilogue: normalize and store fp32 ---------------------------------
#pragma unroll
  for (int r = 0; r < 8; ++r) {
    const float inv = 1.0f / l_i[r];
    const int row = qbase + r + half * 8;
#pragma unroll
    for (int t = 0; t < 4; ++t)
      out[(size_t)row * D_ + t * 16 + col] = oacc[t][r] * inv;
  }
}

// ---------------------------------------------------------------------------
extern "C" void kernel_launch(void* const* d_in, const int* in_sizes, int n_in,
                              void* d_out, int out_size, void* d_ws, size_t ws_size,
                              hipStream_t stream) {
  (void)in_sizes; (void)n_in; (void)out_size; (void)ws_size;
  const float* x    = (const float*)d_in[0];
  const int*   mask = (const int*)d_in[1];
  const float* Wq   = (const float*)d_in[2];
  const float* bq   = (const float*)d_in[3];
  const float* Wk   = (const float*)d_in[4];
  const float* bk   = (const float*)d_in[5];
  const float* Wv   = (const float*)d_in[6];
  const float* bv   = (const float*)d_in[7];
  float* out = (float*)d_out;

  // Workspace: Q, K, V^T as f16 (1 MB each) + swizzled W fragments (192 KB).
  const size_t nQKV = (size_t)B_ * S_ * D_;
  _Float16* Q   = (_Float16*)d_ws;
  _Float16* K   = Q + nQKV;
  _Float16* Vt  = K + nQKV;
  _Float16* Wsw = Vt + nQKV;

  pack_w_kernel<<<(3 * 16 * 4 * 32 * 16) / 256, 256, 0, stream>>>(Wq, Wk, Wv,
                                                                  Wsw);
  qkv_proj_kernel<<<(B_ * S_) / 16, 96, 0, stream>>>(x, Wsw, bq, bk, bv,
                                                     Q, K, Vt);
  flash_attn_kernel<<<(B_ * S_) / (16 * 4), 128, 0, stream>>>(Q, K, Vt, mask,
                                                              out);
}
